// Gamba_17566416240739
// MI455X (gfx1250) — matrix operand order repaired and checked by reading, not compiled, not run on hardware
//
#include <hip/hip_runtime.h>
#include <hip/hip_bf16.h>

typedef __bf16 bf16;
typedef __attribute__((ext_vector_type(16))) __bf16 v16bf;
typedef __attribute__((ext_vector_type(8)))  float  v8f;
typedef unsigned int u32x4 __attribute__((ext_vector_type(4)));
typedef int          i32x8 __attribute__((ext_vector_type(8)));
typedef int          i32x4 __attribute__((ext_vector_type(4)));

#define HDIM 128

// ---------------------------------------------------------------- utilities
__global__ void gamba_zero_f32(float* __restrict__ p, long n) {
  long i = (long)blockIdx.x * blockDim.x + threadIdx.x;
  if (i < n) p[i] = 0.0f;
}

// agg[dst] += x[src] : one wave per edge, 4 floats per lane, L2-resident atomics.
__global__ void gamba_edge_scatter(const float* __restrict__ x,
                                   const int* __restrict__ src,
                                   const int* __restrict__ dst,
                                   float* __restrict__ agg, long E) {
  long idx = (long)blockIdx.x * blockDim.x + threadIdx.x;
  long e = idx >> 5;
  if (e >= E) return;
  int lane = (int)(idx & 31);
  int s = src[e], d = dst[e];
  const float4 v = *(const float4*)(x + (long)s * HDIM + lane * 4);
  float* p = agg + (long)d * HDIM + lane * 4;
  atomicAdd(p + 0, v.x);
  atomicAdd(p + 1, v.y);
  atomicAdd(p + 2, v.z);
  atomicAdd(p + 3, v.w);
}

// ---- TDM descriptor for a [rows x 128] f32 tile of a row-major [*][128] tensor
__device__ __forceinline__ void gamba_tdm_load_tile(const float* gsrc, unsigned lds_addr,
                                                    unsigned rows_left) {
  unsigned long long ga = (unsigned long long)(size_t)gsrc;
  u32x4 g0;
  g0[0] = 1u;                                   // count=1, user-mode descriptor
  g0[1] = lds_addr;                             // LDS byte address
  g0[2] = (unsigned)(ga & 0xffffffffull);       // global_addr[31:0]
  g0[3] = (unsigned)((ga >> 32) & 0x1ffffffull) // global_addr[56:32]
          | (2u << 30);                         // type=2 ("image")
  i32x8 g1;
  g1[0] = (int)(2u << 16);                      // data_size=2 -> 4 bytes
  g1[1] = (int)(128u << 16);                    // tensor_dim0 = 128 (bits 79:48 lo)
  g1[2] = (int)((rows_left & 0xffffu) << 16);   // tensor_dim1 lo16 (bits 111:80)
  g1[3] = (int)((rows_left >> 16) | (128u << 16)); // tensor_dim1 hi16 | tile_dim0=128
  g1[4] = 16;                                   // tile_dim1 = 16, tile_dim2 = 0
  g1[5] = 128;                                  // tensor_dim0_stride = 128 (lo32)
  g1[6] = 0;                                    // stride hi16 | dim1_stride lo16
  g1[7] = 0;
  i32x4 z4 = {0, 0, 0, 0};
  i32x8 z8 = {0, 0, 0, 0, 0, 0, 0, 0};
  __builtin_amdgcn_tensor_load_to_lds(g0, g1, z4, z4, z8, 0);
}

// ------------------------------------------------- WMMA linear: out = (in [+in2]) @ W + b
// W is [128][ncols] row-major with row stride wstride; block handles one 128-col chunk
// (blockIdx.y) and grid-strides over 16-row tiles. 8 waves, wave w owns cols 16w..16w+15.
// A tiles are DMA'd by the Tensor Data Mover into LDS f32 staging, then packed to bf16.
__global__ __launch_bounds__(256) void gamba_linear_wmma(
    const float* __restrict__ in, const float* __restrict__ in2,
    const float* __restrict__ W, int wstride, const float* __restrict__ bias,
    float* __restrict__ out, int nrows, int ncols) {
  __shared__ bf16 WT[128][136];   // [n][k] transposed bf16 weights for this col chunk
  __shared__ bf16 AT[16][136];    // 16-row A tile, bf16
  __shared__ float XS[16][HDIM];  // TDM staging: x tile (f32)
  __shared__ float GS[16][HDIM];  // TDM staging: agg tile (f32)

  const int colchunk = blockIdx.y * 128;
  const int tid  = threadIdx.x;
  const int wave = tid >> 5;
  const int lane = tid & 31;

  for (int e = tid; e < 128 * 128; e += 256) {
    int k = e >> 7, n = e & 127;
    WT[n][k] = (bf16)W[(long)k * wstride + colchunk + n];
  }
  __syncthreads();

  // resident B fragments (4 K-steps of 32) in contiguous K order, same as A:
  // lane l: col n=l&15, K = kb2+0..15, kb2 = (l<16)?0:16, packed pairs per VGPR.
  const int nn  = lane & 15;
  const int kb2 = (lane < 16) ? 0 : 16;
  v16bf bfrag[4];
#pragma unroll
  for (int s = 0; s < 4; ++s)
#pragma unroll
    for (int i = 0; i < 16; ++i)
      bfrag[s][i] = WT[wave * 16 + nn][s * 32 + kb2 + i];

  const float bv = bias ? bias[colchunk + wave * 16 + nn] : 0.0f;
  const int kbA = (lane < 16) ? 0 : 16;
  const unsigned lds_x = (unsigned)(size_t)&XS[0][0];  // LDS aperture: low 32 bits
  const unsigned lds_g = (unsigned)(size_t)&GS[0][0];

  for (int rt = blockIdx.x; rt * 16 < nrows; rt += gridDim.x) {
    const long row0 = (long)rt * 16;
    __syncthreads();
    if (wave == 0) {  // Tensor Data Mover: async DMA of A tile(s) into LDS
      gamba_tdm_load_tile(in + row0 * HDIM, lds_x, (unsigned)(nrows - row0));
      if (in2) gamba_tdm_load_tile(in2 + row0 * HDIM, lds_g, (unsigned)(nrows - row0));
      __builtin_amdgcn_s_wait_tensorcnt((short)0);
    }
    __syncthreads();

    // convert/add/pack: two bf16 per b32 LDS store
    for (int e = tid; e < 16 * 64; e += 256) {
      int r = e >> 6, c2 = (e & 63) * 2;
      float v0 = XS[r][c2], v1 = XS[r][c2 + 1];
      if (in2) { v0 += GS[r][c2]; v1 += GS[r][c2 + 1]; }
      union { bf16 h[2]; unsigned u; } pk;
      pk.h[0] = (bf16)v0;
      pk.h[1] = (bf16)v1;
      *(unsigned*)&AT[r][c2] = pk.u;
    }
    __syncthreads();

    v8f acc;
#pragma unroll
    for (int r = 0; r < 8; ++r) acc[r] = bv;

#pragma unroll
    for (int s = 0; s < 4; ++s) {
      // A fragment: lane l -> row nn, K = s*32 + kbA..kbA+15
      v16bf afrag;
#pragma unroll
      for (int i = 0; i < 16; ++i) {
        afrag[i]     = AT[nn][s * 32 + kbA + i];
      }
      acc = __builtin_amdgcn_wmma_f32_16x16x32_bf16(
          false, afrag, false, bfrag[s], (short)0, acc, false, false);
    }

    // D layout: VGPR r -> row r (lanes 0-15) / r+8 (lanes 16-31), col = lane&15
#pragma unroll
    for (int r = 0; r < 8; ++r) {
      int m = (lane < 16) ? r : (r + 8);
      out[(row0 + m) * (long)ncols + colchunk + wave * 16 + nn] = acc[r];
    }
  }
}

// ----------------------------------------------- attention pool (one block per graph)
// kv: [N][256] (k cols 0..127, v cols 128..255). 8 waves; wave handles 4 (head,token)
// pairs; lane owns keys l, l+32, ..., scores kept in registers, streaming softmax.
__global__ __launch_bounds__(256) void gamba_attn_pool(
    const float* __restrict__ kv, const float* __restrict__ vt,
    const float* __restrict__ qkv_w, const float* __restrict__ qkv_b,
    const float* __restrict__ ao_w, const float* __restrict__ ao_b,
    float* __restrict__ tokens) {
  __shared__ float qs[8][HDIM];
  __shared__ float os[8][HDIM];
  const int g = blockIdx.x;
  const int tid = threadIdx.x, wave = tid >> 5, lane = tid & 31;
  const float scale = 0.17677669529663687f;  // 1/sqrt(32)

  for (int e = tid; e < 8 * HDIM; e += 256) {
    int t = e >> 7, n = e & 127;
    float a = qkv_b[n];
    for (int k = 0; k < HDIM; ++k) a += vt[t * HDIM + k] * qkv_w[k * 384 + n];
    qs[t][n] = a;
    os[t][n] = 0.0f;
  }
  __syncthreads();

  const float* kvg = kv + (long)g * 1024 * 256;
  for (int pp = 0; pp < 4; ++pp) {
    int p = wave * 4 + pp;
    int h = p >> 3, t = p & 7;
    float sc[32];
    float mx = -3.0e38f;
#pragma unroll
    for (int j = 0; j < 32; ++j) {
      const float* krow = kvg + (long)(lane + 32 * j) * 256 + h * 32;
      float s = 0.0f;
#pragma unroll
      for (int d = 0; d < 32; ++d) s += qs[t][h * 32 + d] * krow[d];
      s *= scale;
      sc[j] = s;
      mx = fmaxf(mx, s);
    }
    for (int off = 16; off > 0; off >>= 1) mx = fmaxf(mx, __shfl_xor(mx, off, 32));
    float se = 0.0f;
#pragma unroll
    for (int j = 0; j < 32; ++j) { sc[j] = __expf(sc[j] - mx); se += sc[j]; }
    for (int off = 16; off > 0; off >>= 1) se += __shfl_xor(se, off, 32);
    const float rinv = 1.0f / se;

    float acc[32];
#pragma unroll
    for (int d = 0; d < 32; ++d) acc[d] = 0.0f;
#pragma unroll
    for (int j = 0; j < 32; ++j) {
      const float* vrow = kvg + (long)(lane + 32 * j) * 256 + 128 + h * 32;
      float w = sc[j] * rinv;
#pragma unroll
      for (int d = 0; d < 32; ++d) acc[d] += w * vrow[d];
    }
#pragma unroll
    for (int d = 0; d < 32; ++d) atomicAdd(&os[t][h * 32 + d], acc[d]);
  }
  __syncthreads();

  for (int e = tid; e < 8 * HDIM; e += 256) {
    int t = e >> 7, n = e & 127;
    float a = ao_b[n];
    for (int k = 0; k < HDIM; ++k) a += os[t][k] * ao_w[k * HDIM + n];
    tokens[((long)g * 8 + t) * HDIM + n] = a;
  }
}

// ------------------------------------------------- Mamba block (one block per graph)
__global__ __launch_bounds__(256) void gamba_mamba(
    const float* __restrict__ tokens, const float* __restrict__ in_w,
    const float* __restrict__ conv_w, const float* __restrict__ conv_b,
    const float* __restrict__ x_w, const float* __restrict__ dt_w,
    const float* __restrict__ dt_b, const float* __restrict__ A_log,
    const float* __restrict__ Dw, const float* __restrict__ out_w,
    const float* __restrict__ norm_w, const float* __restrict__ normf_w,
    float* __restrict__ gfm) {
  __shared__ float tok[8][HDIM], hn[8][HDIM];
  __shared__ float uu[8][256], gg[8][256], uc[8][256], dts[8][256], ys[8][256];
  __shared__ float dtr[8][8], Bm[8][16], Cm[8][16];
  const int g = blockIdx.x;
  const int tid = threadIdx.x, wave = tid >> 5, lane = tid & 31;

  for (int e = tid; e < 8 * HDIM; e += 256)
    tok[e >> 7][e & 127] = tokens[(long)g * 8 * HDIM + e];
  __syncthreads();

  {  // rmsnorm: wave w handles row w
    int t = wave;
    float ss = 0.0f;
#pragma unroll
    for (int i = 0; i < 4; ++i) { float v = tok[t][lane * 4 + i]; ss += v * v; }
    for (int off = 16; off > 0; off >>= 1) ss += __shfl_xor(ss, off, 32);
    float rn = rsqrtf(ss * (1.0f / HDIM) + 1e-5f);
#pragma unroll
    for (int i = 0; i < 4; ++i) {
      int c = lane * 4 + i;
      hn[t][c] = tok[t][c] * rn * norm_w[c];
    }
  }
  __syncthreads();

  for (int e = tid; e < 8 * 512; e += 256) {  // in-proj -> u | gate
    int t = e >> 9, j = e & 511;
    float a = 0.0f;
    for (int k = 0; k < HDIM; ++k) a += hn[t][k] * in_w[k * 512 + j];
    if (j < 256) uu[t][j] = a; else gg[t][j - 256] = a;
  }
  __syncthreads();

  for (int e = tid; e < 8 * 256; e += 256) {  // causal depthwise conv + silu
    int t = e >> 8, i = e & 255;
    float a = conv_b[i];
#pragma unroll
    for (int k = 0; k < 4; ++k) {
      int tt = t + k - 3;
      if (tt >= 0) a += conv_w[i * 4 + k] * uu[tt][i];
    }
    uc[t][i] = a / (1.0f + __expf(-a));
  }
  __syncthreads();

  for (int e = tid; e < 8 * 40; e += 256) {  // ssm proj -> dtr | B | C
    int t = e / 40, j = e % 40;
    float a = 0.0f;
    for (int k = 0; k < 256; ++k) a += uc[t][k] * x_w[k * 40 + j];
    if (j < 8) dtr[t][j] = a;
    else if (j < 24) Bm[t][j - 8] = a;
    else Cm[t][j - 24] = a;
  }
  __syncthreads();

  for (int e = tid; e < 8 * 256; e += 256) {  // dt = softplus(dtr @ dt_w + b)
    int t = e >> 8, i = e & 255;
    float a = dt_b[i];
#pragma unroll
    for (int r = 0; r < 8; ++r) a += dtr[t][r] * dt_w[r * 256 + i];
    dts[t][i] = (a > 20.0f) ? a : log1pf(__expf(a));
  }
  __syncthreads();

  {  // selective scan: thread = channel
    const int i = tid;
    float hst[16], Ai[16];
#pragma unroll
    for (int s = 0; s < 16; ++s) { hst[s] = 0.0f; Ai[s] = -__expf(A_log[i * 16 + s]); }
    const float Dv = Dw[i];
    for (int t = 0; t < 8; ++t) {
      float dti = dts[t][i], ui = uc[t][i], yv = 0.0f;
#pragma unroll
      for (int s = 0; s < 16; ++s) {
        float dA = __expf(dti * Ai[s]);
        hst[s] = dA * hst[s] + dti * Bm[t][s] * ui;
        yv += hst[s] * Cm[t][s];
      }
      yv += ui * Dv;
      float gt = gg[t][i];
      ys[t][i] = yv * (gt / (1.0f + __expf(-gt)));
    }
  }
  __syncthreads();

  for (int e = tid; e < 8 * HDIM; e += 256) {  // out proj + residual (into hn)
    int t = e >> 7, c = e & 127;
    float a = 0.0f;
    for (int i = 0; i < 256; ++i) a += ys[t][i] * out_w[i * HDIM + c];
    hn[t][c] = tok[t][c] + a;
  }
  __syncthreads();

  {  // final rmsnorm in place
    int t = wave;
    float ss = 0.0f;
#pragma unroll
    for (int i = 0; i < 4; ++i) { float v = hn[t][lane * 4 + i]; ss += v * v; }
    for (int off = 16; off > 0; off >>= 1) ss += __shfl_xor(ss, off, 32);
    float rn = rsqrtf(ss * (1.0f / HDIM) + 1e-5f);
#pragma unroll
    for (int i = 0; i < 4; ++i) {
      int c = lane * 4 + i;
      hn[t][c] = hn[t][c] * rn * normf_w[c];
    }
  }
  __syncthreads();

  if (tid < HDIM) {  // mean over T
    float a = 0.0f;
#pragma unroll
    for (int t = 0; t < 8; ++t) a += hn[t][tid];
    gfm[(long)g * HDIM + tid] = a * 0.125f;
  }
}

// x = g + gf_mean[graph]   (graph = row >> 10, batch is contiguous)
__global__ void gamba_x_update(float* __restrict__ x, const float* __restrict__ g,
                               const float* __restrict__ gfm, long n) {
  long i = (long)blockIdx.x * blockDim.x + threadIdx.x;
  if (i >= n) return;
  long row = i >> 7;
  x[i] = g[i] + gfm[(row >> 10) * HDIM + (int)(i & 127)];
}

// final segment sum: one block per graph
__global__ void gamba_seg_reduce(const float* __restrict__ y, float* __restrict__ out) {
  __shared__ float part[2][HDIM];
  const int g = blockIdx.x, tid = threadIdx.x;
  const int c = tid & 127, half = tid >> 7;
  const float* base = y + ((long)g * 1024 + half * 512) * HDIM;
  float acc = 0.0f;
  for (int r = 0; r < 512; ++r) acc += base[(long)r * HDIM + c];
  part[half][c] = acc;
  __syncthreads();
  if (tid < HDIM) out[(long)g * HDIM + tid] = part[0][tid] + part[1][tid];
}

// ---------------------------------------------------------------- launcher
extern "C" void kernel_launch(void* const* d_in, const int* in_sizes, int n_in,
                              void* d_out, int out_size, void* d_ws, size_t ws_size,
                              hipStream_t stream) {
  const float* x_in     = (const float*)d_in[0];
  const int*   eidx     = (const int*)d_in[1];
  const float* w_in     = (const float*)d_in[5];
  const float* b_in     = (const float*)d_in[6];
  const float* gin_w    = (const float*)d_in[7];
  const float* gin_b    = (const float*)d_in[8];
  const float* vt       = (const float*)d_in[9];
  const float* qkv_w    = (const float*)d_in[10];
  const float* qkv_b    = (const float*)d_in[11];
  const float* ao_w     = (const float*)d_in[12];
  const float* ao_b     = (const float*)d_in[13];
  const float* m_in_w   = (const float*)d_in[14];
  const float* m_conv_w = (const float*)d_in[15];
  const float* m_conv_b = (const float*)d_in[16];
  const float* m_x_w    = (const float*)d_in[17];
  const float* m_dt_w   = (const float*)d_in[18];
  const float* m_dt_b   = (const float*)d_in[19];
  const float* m_A_log  = (const float*)d_in[20];
  const float* m_D      = (const float*)d_in[21];
  const float* m_out_w  = (const float*)d_in[22];
  const float* m_norm_w = (const float*)d_in[23];
  const float* m_normf_w= (const float*)d_in[24];
  const float* w_out    = (const float*)d_in[25];
  const float* b_out    = (const float*)d_in[26];

  const long N = in_sizes[0] / HDIM;
  const long E = in_sizes[1] / 2;
  const int  L = in_sizes[7] / (HDIM * HDIM);
  const int  B = (int)(N >> 10);  // nodes_per_graph = 1024 (contiguous batch)

  float* ws     = (float*)d_ws;
  float* xcur   = ws;                       // N*128
  float* gbuf   = xcur + N * HDIM;          // N*128
  float* aggkv  = gbuf + N * HDIM;          // N*256 : agg (first half) / kv, disjoint lifetimes
  float* tokens = aggkv + N * 256;          // B*8*128
  float* gfm    = tokens + (long)B * 8 * HDIM;  // B*128

  const int* src = eidx;
  const int* dst = eidx + E;

  const dim3 blk(256);
  const long nScatterBlk = (E * 32 + 255) / 256;
  const long nElemBlk    = (N * HDIM + 255) / 256;
  const dim3 gemmGrid(2048, 1);
  const dim3 gemmGridKV(2048, 2);

  auto gin = [&](const float* in, const float* W, const float* bias, float* out) {
    gamba_zero_f32<<<nElemBlk, blk, 0, stream>>>(aggkv, N * HDIM);
    gamba_edge_scatter<<<nScatterBlk, blk, 0, stream>>>(in, src, dst, aggkv, E);
    gamba_linear_wmma<<<gemmGrid, blk, 0, stream>>>(in, aggkv, W, HDIM, bias, out,
                                                    (int)N, HDIM);
  };

  gin(x_in, w_in, b_in, xcur);
  for (int l = 0; l < L; ++l) {
    gin(xcur, gin_w + (long)l * HDIM * HDIM, gin_b + (long)l * HDIM, gbuf);
    // K|V projection: qkv_w columns 128..383, row stride 384
    gamba_linear_wmma<<<gemmGridKV, blk, 0, stream>>>(
        xcur, nullptr, qkv_w + (long)l * HDIM * 384 + HDIM, 384,
        qkv_b + (long)l * 384 + HDIM, aggkv, (int)N, 256);
    gamba_attn_pool<<<B, blk, 0, stream>>>(
        aggkv, vt + (long)l * 8 * HDIM, qkv_w + (long)l * HDIM * 384,
        qkv_b + (long)l * 384, ao_w + (long)l * HDIM * HDIM,
        ao_b + (long)l * HDIM, tokens);
    gamba_mamba<<<B, blk, 0, stream>>>(tokens, m_in_w, m_conv_w, m_conv_b, m_x_w,
                                       m_dt_w, m_dt_b, m_A_log, m_D, m_out_w,
                                       m_norm_w, m_normf_w, gfm);
    gamba_x_update<<<nElemBlk, blk, 0, stream>>>(xcur, gbuf, gfm, N * HDIM);
  }
  gin(xcur, w_out, b_out, gbuf);
  gamba_seg_reduce<<<B, blk, 0, stream>>>(gbuf, (float*)d_out);
}
